// Net_68015102099873
// MI455X (gfx1250) — matrix operand order repaired and checked by reading, not compiled, not run on hardware
//
#include <hip/hip_runtime.h>
#include <hip/hip_bf16.h>
#include <math.h>

typedef __attribute__((ext_vector_type(16))) __bf16 v16bf;
typedef __attribute__((ext_vector_type(8)))  float  v8f;

#define RDIM 2240   // RTN + RAN + ME = 128 + 64 + 2048
#define TIN  67712  // ME + EAN + MR + RAN

// pack 16 f32 (4x float4) -> v16bf with RNE conversion (v_cvt_pk_bf16_f32)
__device__ __forceinline__ v16bf pack_bf16(float4 a, float4 b, float4 c, float4 d) {
    v16bf v;
    v[0]  = (__bf16)a.x; v[1]  = (__bf16)a.y; v[2]  = (__bf16)a.z; v[3]  = (__bf16)a.w;
    v[4]  = (__bf16)b.x; v[5]  = (__bf16)b.y; v[6]  = (__bf16)b.z; v[7]  = (__bf16)b.w;
    v[8]  = (__bf16)c.x; v[9]  = (__bf16)c.y; v[10] = (__bf16)c.z; v[11] = (__bf16)c.w;
    v[12] = (__bf16)d.x; v[13] = (__bf16)d.y; v[14] = (__bf16)d.z; v[15] = (__bf16)d.w;
    return v;
}

// ---------------- init: eg[64], rg[128] = -inf ----------------
__global__ void __launch_bounds__(256) init_kernel(float* ws) {
    int t = threadIdx.x;
    if (t < 192) ws[t] = -INFINITY;   // [0,64)=eg, [64,192)=rg
}

// ---------------- entity encoder + segment max ----------------
__global__ void __launch_bounds__(256) entity_kernel(
    const float* __restrict__ entity,
    const float* __restrict__ ew1, const float* __restrict__ eb1,
    const float* __restrict__ ew2, const float* __restrict__ eb2,
    const float* __restrict__ ew3, const float* __restrict__ eb3,
    float* __restrict__ eg)
{
    __shared__ float w1[32 * 128];
    int t = threadIdx.x;
    for (int i = t; i < 32 * 128; i += 256) w1[i] = ew1[i];
    __syncthreads();

    int e = blockIdx.x * 256 + t;                  // 0..2047
    const float* row = entity + (size_t)e * 128;

    float a1[32];
    for (int n = 0; n < 32; ++n) a1[n] = eb1[n];
    for (int k = 0; k < 128; ++k) {
        float x = row[k];
        for (int n = 0; n < 32; ++n) a1[n] += x * w1[n * 128 + k];
    }
    for (int n = 0; n < 32; ++n) a1[n] = fmaxf(a1[n], 0.0f);

    float a2[8];
    for (int n = 0; n < 8; ++n) {
        float acc = eb2[n];
        for (int k = 0; k < 32; ++k) acc += a1[k] * ew2[n * 32 + k];
        a2[n] = fmaxf(acc, 0.0f);
    }
    float v = eb3[0];
    for (int k = 0; k < 8; ++k) v += a2[k] * ew3[k];

    // one-hot type in cols [0,64)
    for (int c = 0; c < 64; ++c) {
        if (row[c] == 1.0f) { atomicMax(&eg[c], v); break; }
    }
}

// ------------- relationship encoder (WMMA bf16) + segment max -------------
__global__ void __launch_bounds__(256) rel_kernel(
    const float* __restrict__ rel,
    const float* __restrict__ rw1, const float* __restrict__ rb1,
    const float* __restrict__ rw2, const float* __restrict__ rb2,
    const float* __restrict__ rw3, const float* __restrict__ rb3,
    const float* __restrict__ rw4, const float* __restrict__ rb4,
    const float* __restrict__ rw5, const float* __restrict__ rb5,
    float* __restrict__ rg)
{
    __shared__ float h1[16 * 128];
    __shared__ float h2[16 * 64];
    __shared__ float h3[16 * 32];
    __shared__ float h4[16 * 8];
    __shared__ float rfv[16];

    const int t    = threadIdx.x;
    const int wave = t >> 5;
    const int lane = t & 31;
    const int lm   = lane & 15;        // row / col within tile
    const int hi   = lane >> 4;        // K-half selector
    const int m0   = blockIdx.x * 16;  // 16 relationship rows per block
    const int n    = wave * 16 + lm;   // this lane's output column (0..127)

    // per-lane base pointers (A layout §7.12.2: lanes 0-15 K {0..7,16..23},
// lanes 16-31 K {8..15,24..31}; B mirror: same K sets as A)
    const float* __restrict__ aP0 = rel + (size_t)(m0 + lm) * RDIM + hi * 8;
    const float* __restrict__ aP1 = aP0 + 16;
    const float* __restrict__ bP  = rw1 + (size_t)n * RDIM + hi * 8;

    v8f c0 = {}, c1 = {};
    for (int k0 = 0; k0 < RDIM; k0 += 64) {
        // issue ALL raw loads for both 32-K chunks before consuming anything,
        // so chunk1 loads stay in flight during chunk0's convert+WMMA.
        const float4* pa0 = (const float4*)(aP0 + k0);
        const float4* pa1 = (const float4*)(aP1 + k0);
        const float4* pb0 = (const float4*)(bP  + k0);
        const float4* pa2 = (const float4*)(aP0 + k0 + 32);
        const float4* pa3 = (const float4*)(aP1 + k0 + 32);
        const float4* pb1 = (const float4*)(bP  + k0 + 32);

        float4 A0 = pa0[0], A1 = pa0[1], A2 = pa1[0], A3 = pa1[1];
        float4 B0 = pb0[0], B1 = pb0[1], B2 = pb0[4], B3 = pb0[5];
        float4 A4 = pa2[0], A5 = pa2[1], A6 = pa3[0], A7 = pa3[1];
        float4 B4 = pb1[0], B5 = pb1[1], B6 = pb1[4], B7 = pb1[5];

        if (k0 + 64 < RDIM) {
            __builtin_prefetch(aP0 + k0 + 64, 0, 1);
            __builtin_prefetch(bP  + k0 + 64, 0, 1);
        }

        // chunk 0
        {
            v16bf a = pack_bf16(A0, A1, A2, A3);
            v16bf b = pack_bf16(B0, B1, B2, B3);
            c0 = __builtin_amdgcn_wmma_f32_16x16x32_bf16(
                     false, a, false, b, (short)0, c0, false, false);
        }
        // chunk 1 (independent accumulator breaks the C dependency chain)
        {
            v16bf a = pack_bf16(A4, A5, A6, A7);
            v16bf b = pack_bf16(B4, B5, B6, B7);
            c1 = __builtin_amdgcn_wmma_f32_16x16x32_bf16(
                     false, a, false, b, (short)0, c1, false, false);
        }
    }

    // epilogue: combine accumulators, bias + relu, scatter C tile to LDS h1[m][n]
    {
        float bias = rb1[n];
        for (int r = 0; r < 8; ++r) {
            int m = r + 8 * hi;
            h1[m * 128 + n] = fmaxf(c0[r] + c1[r] + bias, 0.0f);
        }
    }
    __syncthreads();

    // layer 2: [16,128] x [128,64] -> h2
    {
        int m = t >> 4, nc = t & 15;
        for (int rep = 0; rep < 4; ++rep) {
            int n2 = nc + 16 * rep;
            float acc = rb2[n2];
            const float* w = rw2 + n2 * 128;
            for (int k = 0; k < 128; ++k) acc += h1[m * 128 + k] * w[k];
            h2[m * 64 + n2] = fmaxf(acc, 0.0f);
        }
    }
    __syncthreads();

    // layer 3: [16,64] x [64,32] -> h3
    {
        int m = t >> 4, nc = t & 15;
        for (int rep = 0; rep < 2; ++rep) {
            int n3 = nc + 16 * rep;
            float acc = rb3[n3];
            const float* w = rw3 + n3 * 64;
            for (int k = 0; k < 64; ++k) acc += h2[m * 64 + k] * w[k];
            h3[m * 32 + n3] = fmaxf(acc, 0.0f);
        }
    }
    __syncthreads();

    // layer 4: [16,32] x [32,8] -> h4
    if (t < 128) {
        int m = t >> 3, n4 = t & 7;
        float acc = rb4[n4];
        const float* w = rw4 + n4 * 32;
        for (int k = 0; k < 32; ++k) acc += h3[m * 32 + k] * w[k];
        h4[m * 8 + n4] = fmaxf(acc, 0.0f);
    }
    __syncthreads();

    // layer 5: [16,8] x [8,1] -> rfv (no relu)
    if (t < 16) {
        float acc = rb5[0];
        for (int k = 0; k < 8; ++k) acc += h4[t * 8 + k] * rw5[k];
        rfv[t] = acc;
    }
    __syncthreads();

    // one-hot masked segment max over the 128 type columns
    if (t < 128) {
        float best = -INFINITY;
        bool found = false;
        for (int m = 0; m < 16; ++m) {
            if (rel[(size_t)(m0 + m) * RDIM + t] == 1.0f) {
                found = true;
                best = fmaxf(best, rfv[m]);
            }
        }
        if (found) atomicMax(&rg[t], best);
    }
}

// ---------------- target encoder layer 1: 64 long dot products ----------------
__global__ void __launch_bounds__(256) target1_kernel(
    const float* __restrict__ target, const float* __restrict__ tw1,
    const float* __restrict__ tb1, float* __restrict__ tf1)
{
    int b = blockIdx.x, t = threadIdx.x;
    const float* w = tw1 + (size_t)b * TIN;
    float s = 0.0f;
    for (int i = t; i < TIN; i += 256) s += target[i] * w[i];
    __shared__ float red[256];
    red[t] = s;
    __syncthreads();
    for (int k = 128; k > 0; k >>= 1) {
        if (t < k) red[t] += red[t + k];
        __syncthreads();
    }
    if (t == 0) tf1[b] = fmaxf(red[0] + tb1[b], 0.0f);
}

// ---------------- target layer2 + GRU + softmax head ----------------
__global__ void __launch_bounds__(256) final_kernel(
    const float* __restrict__ tw2, const float* __restrict__ tb2,
    const float* __restrict__ wih, const float* __restrict__ whh,
    const float* __restrict__ bih, const float* __restrict__ bhh,
    const float* __restrict__ hidden,
    const float* __restrict__ aw,  const float* __restrict__ ab,
    const float* __restrict__ action_mask,
    const float* __restrict__ eg, const float* __restrict__ rg,
    const float* __restrict__ tf1, float* __restrict__ out)
{
    __shared__ float env[208];
    __shared__ float h[128];
    __shared__ float red[256];
    int t = threadIdx.x;

    if (t < 64) {
        float v = eg[t];
        env[t] = (v == -INFINITY) ? 0.0f : v;       // empty group -> 0
    } else if (t < 192) {
        float v = rg[t - 64];
        env[t] = (v == -INFINITY) ? 0.0f : v;
    }
    if (t < 16) {
        float acc = tb2[t];
        for (int k = 0; k < 64; ++k) acc += tf1[k] * tw2[t * 64 + k];
        env[192 + t] = acc;
    }
    __syncthreads();

    if (t < 128) {
        float xr = bih[t], xz = bih[128 + t], xn = bih[256 + t];
        for (int k = 0; k < 208; ++k) {
            float e = env[k];
            xr += wih[t * 208 + k] * e;
            xz += wih[(128 + t) * 208 + k] * e;
            xn += wih[(256 + t) * 208 + k] * e;
        }
        float hr = bhh[t], hz = bhh[128 + t], hn = bhh[256 + t];
        for (int k = 0; k < 128; ++k) {
            float hv = hidden[k];
            hr += whh[t * 128 + k] * hv;
            hz += whh[(128 + t) * 128 + k] * hv;
            hn += whh[(256 + t) * 128 + k] * hv;
        }
        float r = 1.0f / (1.0f + expf(-(xr + hr)));
        float z = 1.0f / (1.0f + expf(-(xz + hz)));
        float nn = tanhf(xn + r * hn);
        h[t] = (1.0f - z) * nn + z * hidden[t];
    }
    __syncthreads();

    float logit = ab[t];
    for (int k = 0; k < 128; ++k) logit += aw[t * 128 + k] * h[k];

    // softmax over 256
    red[t] = logit;
    __syncthreads();
    for (int s = 128; s > 0; s >>= 1) {
        if (t < s) red[t] = fmaxf(red[t], red[t + s]);
        __syncthreads();
    }
    float mx = red[0];
    __syncthreads();
    float e = expf(logit - mx);
    red[t] = e;
    __syncthreads();
    for (int s = 128; s > 0; s >>= 1) {
        if (t < s) red[t] += red[t + s];
        __syncthreads();
    }
    float sum = red[0];
    out[t] = (e / sum) * action_mask[t];
}

extern "C" void kernel_launch(void* const* d_in, const int* in_sizes, int n_in,
                              void* d_out, int out_size, void* d_ws, size_t ws_size,
                              hipStream_t stream) {
    const float* entity       = (const float*)d_in[0];
    const float* relationship = (const float*)d_in[1];
    const float* target       = (const float*)d_in[2];
    const float* action_mask  = (const float*)d_in[3];
    const float* ew1 = (const float*)d_in[4];  const float* eb1 = (const float*)d_in[5];
    const float* ew2 = (const float*)d_in[6];  const float* eb2 = (const float*)d_in[7];
    const float* ew3 = (const float*)d_in[8];  const float* eb3 = (const float*)d_in[9];
    const float* rw1 = (const float*)d_in[10]; const float* rb1 = (const float*)d_in[11];
    const float* rw2 = (const float*)d_in[12]; const float* rb2 = (const float*)d_in[13];
    const float* rw3 = (const float*)d_in[14]; const float* rb3 = (const float*)d_in[15];
    const float* rw4 = (const float*)d_in[16]; const float* rb4 = (const float*)d_in[17];
    const float* rw5 = (const float*)d_in[18]; const float* rb5 = (const float*)d_in[19];
    const float* tw1 = (const float*)d_in[20]; const float* tb1 = (const float*)d_in[21];
    const float* tw2 = (const float*)d_in[22]; const float* tb2 = (const float*)d_in[23];
    const float* wih = (const float*)d_in[24]; const float* whh = (const float*)d_in[25];
    const float* bih = (const float*)d_in[26]; const float* bhh = (const float*)d_in[27];
    const float* hidden = (const float*)d_in[28];
    const float* aw  = (const float*)d_in[29]; const float* ab  = (const float*)d_in[30];

    float* ws  = (float*)d_ws;
    float* eg  = ws;         // 64
    float* rg  = ws + 64;    // 128
    float* tf1 = ws + 192;   // 64

    init_kernel<<<1, 256, 0, stream>>>(ws);
    entity_kernel<<<8, 256, 0, stream>>>(entity, ew1, eb1, ew2, eb2, ew3, eb3, eg);
    rel_kernel<<<4096, 256, 0, stream>>>(relationship, rw1, rb1, rw2, rb2,
                                         rw3, rb3, rw4, rb4, rw5, rb5, rg);
    target1_kernel<<<64, 256, 0, stream>>>(target, tw1, tb1, tf1);
    final_kernel<<<1, 256, 0, stream>>>(tw2, tb2, wih, whh, bih, bhh, hidden,
                                        aw, ab, action_mask, eg, rg, tf1,
                                        (float*)d_out);
}
